// MultiHeadSelfAttention_26190710571733
// MI455X (gfx1250) — matrix operand order repaired and checked by reading, not compiled, not run on hardware
//
#include <hip/hip_runtime.h>
#include <hip/hip_bf16.h>

// ---------------------------------------------------------------------------
// MI455X (gfx1250) multi-head self-attention, bf16 WMMA path.
//   qkv = x @ W_qkv + b ; flash attention per (b,h) ; out = O @ W_proj + b
// Matrix math: v_wmma_f32_16x16x32_bf16 (wave32 fragments).
// K/V tiles staged via async global->LDS DMA (ASYNCcnt), double-buffered.
// Softmax denominator computed on the matrix core with a ones-fragment.
// ---------------------------------------------------------------------------

typedef __attribute__((ext_vector_type(16))) __bf16 v16bf;
typedef __attribute__((ext_vector_type(8)))  float  v8f;

union Frag { v16bf v; uint4 q[2]; };

__device__ __forceinline__ v8f wmma_bf16(v16bf a, v16bf b, v8f c) {
    // D = A(16x32) * B(32x16) + C, f32 accumulate
    return __builtin_amdgcn_wmma_f32_16x16x32_bf16(
        /*neg_a=*/false, a, /*neg_b=*/false, b,
        /*c_mod=*/(short)0, c, /*reuse_a=*/false, /*reuse_b=*/false);
}

// A fragment from a row-major matrix: lane holds row (lane&15); K elements
// live in two contiguous 16B chunks: [k0+8h .. +7] and [k0+16+8h .. +7].
__device__ __forceinline__ void load_a_frag(Frag& f, const __bf16* rowPtr,
                                            int k0, int half) {
    f.q[0] = *(const uint4*)(rowPtr + k0 + 8 * half);
    f.q[1] = *(const uint4*)(rowPtr + k0 + 16 + 8 * half);
}

// B fragment where the B matrix is stored TRANSPOSED row-major (rows = N,
// cols = K). Lane holds col (lane&15) = a row of B^T; its 16 reduction
// elements live in two 16B chunks: [k0+8h .. +7] and [k0+16+8h .. +7].
__device__ __forceinline__ void load_b_fragT(Frag& f, const __bf16* rowPtr,
                                             int k0, int half) {
    f.q[0] = *(const uint4*)(rowPtr + k0 + 8 * half);
    f.q[1] = *(const uint4*)(rowPtr + k0 + 16 + 8 * half);
}

// ---- CDNA5 async global->LDS copy (16B per lane), ASYNCcnt-tracked --------
// LDS address = low 32 bits of the generic pointer (wave-relative LDS offset).
__device__ __forceinline__ void async_cp16(const __bf16* gsrc, __bf16* ldst) {
    unsigned lds_off = (unsigned)(uintptr_t)ldst;
    unsigned long long gaddr = (unsigned long long)(uintptr_t)gsrc;
    asm volatile("global_load_async_to_lds_b128 %0, %1, off"
                 :: "v"(lds_off), "v"(gaddr)
                 : "memory");
}
__device__ __forceinline__ void wait_async0() {
    asm volatile("s_wait_asynccnt 0x0" ::: "memory");
}

// ---------------------------------------------------------------------------
// Prep kernels: fp32 -> bf16 convert, and transpose+convert for weights.
// ---------------------------------------------------------------------------
__global__ void cvt_bf16_kernel(const float* __restrict__ src,
                                __bf16* __restrict__ dst, int n) {
    int i = blockIdx.x * blockDim.x + threadIdx.x;
    if (i < n) dst[i] = (__bf16)src[i];
}

// W: (K x N) row-major fp32  ->  WT: (N x K) row-major bf16
__global__ void transpose_cvt_kernel(const float* __restrict__ W,
                                     __bf16* __restrict__ WT, int K, int N) {
    int i = blockIdx.x * blockDim.x + threadIdx.x;
    if (i < K * N) {
        int k = i / N, n = i - k * N;
        WT[(size_t)n * K + k] = (__bf16)W[i];
    }
}

// ---------------------------------------------------------------------------
// Tiled bf16 WMMA GEMM:  C(MxN) = A(MxK) @ B(KxN) + bias, B given transposed.
// Block = 256 threads = 8 waves, arranged 4(M) x 2(N); wave tile 32x64.
// MODE 0: write fp32 C (+bias) row-major.
// MODE 1: QKV epilogue -> scatter bf16 Q (scaled 1/8), K, V^T per head.
// ---------------------------------------------------------------------------
template <int MODE>
__global__ __launch_bounds__(256) void gemm_bf16_wmma(
    const __bf16* __restrict__ A, const __bf16* __restrict__ BT,
    const float* __restrict__ bias, float* __restrict__ C,
    __bf16* __restrict__ Qb, __bf16* __restrict__ Kb,
    __bf16* __restrict__ Vtb, int M, int N, int K) {
    const int lane = threadIdx.x & 31;
    const int w    = threadIdx.x >> 5;
    const int half = lane >> 4;
    const int ln   = lane & 15;
    const int wm = w & 3, wn = w >> 2;
    const int mBase = blockIdx.y * 128 + wm * 32;
    const int nBase = blockIdx.x * 128 + wn * 64;

    const __bf16* aRow[2];
    aRow[0] = A + (size_t)(mBase + ln) * K;
    aRow[1] = A + (size_t)(mBase + 16 + ln) * K;
    const __bf16* bRow[4];
#pragma unroll
    for (int j = 0; j < 4; ++j) bRow[j] = BT + (size_t)(nBase + j * 16 + ln) * K;

    v8f acc[2][4] = {};

    for (int k0 = 0; k0 < K; k0 += 32) {
        Frag a[2], b[4];
#pragma unroll
        for (int i = 0; i < 2; ++i) load_a_frag(a[i], aRow[i], k0, half);
#pragma unroll
        for (int j = 0; j < 4; ++j) load_b_fragT(b[j], bRow[j], k0, half);
        // Pull the next K-slab toward the caches (global_prefetch_b8).
        __builtin_prefetch(aRow[0] + k0 + 128, 0, 1);
        __builtin_prefetch(bRow[0] + k0 + 128, 0, 1);
#pragma unroll
        for (int i = 0; i < 2; ++i)
#pragma unroll
            for (int j = 0; j < 4; ++j)
                acc[i][j] = wmma_bf16(a[i].v, b[j].v, acc[i][j]);
    }

#pragma unroll
    for (int i = 0; i < 2; ++i) {
#pragma unroll
        for (int j = 0; j < 4; ++j) {
            const int col = nBase + j * 16 + ln;
            const float bv = bias[col];
#pragma unroll
            for (int r = 0; r < 8; ++r) {
                const int row = mBase + i * 16 + r + 8 * half;  // C layout
                const float val = acc[i][j][r] + bv;
                if (MODE == 0) {
                    C[(size_t)row * N + col] = val;
                } else {
                    // col = three*1024 + h*64 + d ; row = b*2048 + n
                    const int three = col >> 10;
                    const int rem = col & 1023;
                    const int h = rem >> 6;
                    const int d = rem & 63;
                    const int bb = row >> 11;
                    const int n = row & 2047;
                    const int bh = (bb << 4) + h;
                    if (three == 0)
                        Qb[((size_t)bh * 2048 + n) * 64 + d] =
                            (__bf16)(val * 0.125f);  // fold SCALE = Hd^-0.5
                    else if (three == 1)
                        Kb[((size_t)bh * 2048 + n) * 64 + d] = (__bf16)val;
                    else
                        Vtb[((size_t)bh * 64 + d) * 2048 + n] = (__bf16)val;
                }
            }
        }
    }
}

// ---------------------------------------------------------------------------
// Flash attention: grid = (B*H)*16 blocks; block = 8 waves; each wave owns
// 16 query rows. K and V^T tiles (64 keys) are DMA'd into LDS with
// global_load_async_to_lds_b128, double-buffered so the copy of block i+1
// overlaps the 16 WMMAs + softmax of block i. Row-max via batched shfl_xor;
// row-sum (softmax denominator) via an extra WMMA against a ones-fragment.
// ---------------------------------------------------------------------------
__global__ __launch_bounds__(256) void flash_attn_wmma(
    const __bf16* __restrict__ Qb, const __bf16* __restrict__ Kb,
    const __bf16* __restrict__ Vtb, __bf16* __restrict__ Ob) {
    // 72-element row stride (144 B) -> lane stride 36 dwords: conflict-free b128
    __shared__ __align__(16) __bf16 Klds[2][64][72];
    __shared__ __align__(16) __bf16 Vlds[2][64][72];
    __shared__ __align__(16) __bf16 Plds[8][16][72];

    const int tid  = threadIdx.x;
    const int lane = tid & 31;
    const int w    = tid >> 5;
    const int half = lane >> 4;
    const int ln   = lane & 15;
    const int bh   = blockIdx.x >> 4;
    const int qblk = blockIdx.x & 15;
    const int qbase = qblk * 128 + w * 16;

    const __bf16* Q  = Qb + (size_t)bh * 2048 * 64;
    const __bf16* Km = Kb + (size_t)bh * 2048 * 64;
    const __bf16* Vt = Vtb + (size_t)bh * 64 * 2048;

    Frag qf[2];
    const __bf16* qrow = Q + (size_t)(qbase + ln) * 64;
    load_a_frag(qf[0], qrow, 0, half);
    load_a_frag(qf[1], qrow, 32, half);

    Frag ones;  // bf16 1.0 = 0x3F80 -> row-sum B fragment
    ones.q[0] = make_uint4(0x3f803f80u, 0x3f803f80u, 0x3f803f80u, 0x3f803f80u);
    ones.q[1] = ones.q[0];

    float mst[8];
#pragma unroll
    for (int r = 0; r < 8; ++r) mst[r] = -3.0e38f;
    v8f lacc = {};   // softmax denominator, same recurrence as o
    v8f o[4] = {};

    // Cooperative async DMA of one 64-key K tile + V^T tile into buffer `buf`.
    // 512 16B chunks per tile, 256 threads -> 2 chunks each per tile.
    auto issue_tiles = [&](int key0, int buf) {
#pragma unroll
        for (int c = tid; c < 512; c += 256) {
            const int row = c >> 3;            // 0..63
            const int col = (c & 7) * 8;       // element offset, 16B chunks
            async_cp16(Km + (size_t)(key0 + row) * 64 + col,
                       &Klds[buf][row][col]);
            async_cp16(Vt + (size_t)row * 2048 + key0 + col,
                       &Vlds[buf][row][col]);
        }
    };

    issue_tiles(0, 0);

    for (int it = 0; it < 32; ++it) {
        const int buf = it & 1;
        wait_async0();        // own async copies done
        __syncthreads();      // everyone's copies done; prev compute finished
        if (it + 1 < 32) issue_tiles((it + 1) * 64, buf ^ 1);

        // ---- S = Qs @ K^T  (16 q-rows x 64 keys) ----
        v8f s[4] = {};
#pragma unroll
        for (int t = 0; t < 4; ++t) {
            const __bf16* krow = &Klds[buf][t * 16 + ln][0];
            Frag kb0, kb1;
            load_b_fragT(kb0, krow, 0, half);
            load_b_fragT(kb1, krow, 32, half);
            s[t] = wmma_bf16(qf[0].v, kb0.v, s[t]);
            s[t] = wmma_bf16(qf[1].v, kb1.v, s[t]);
        }

        // ---- row max: batch 8 independent bpermutes per xor step ----
        float mx[8];
#pragma unroll
        for (int r = 0; r < 8; ++r)
            mx[r] = fmaxf(fmaxf(s[0][r], s[1][r]), fmaxf(s[2][r], s[3][r]));
#pragma unroll
        for (int off = 8; off >= 1; off >>= 1) {
            float tswp[8];
#pragma unroll
            for (int r = 0; r < 8; ++r) tswp[r] = __shfl_xor(mx[r], off, 32);
#pragma unroll
            for (int r = 0; r < 8; ++r) mx[r] = fmaxf(mx[r], tswp[r]);
        }

        // ---- rescale state, exponentiate ----
#pragma unroll
        for (int r = 0; r < 8; ++r) {
            const float mnew = fmaxf(mst[r], mx[r]);
            const float alpha = __expf(mst[r] - mnew);
            mst[r] = mnew;
#pragma unroll
            for (int t = 0; t < 4; ++t) s[t][r] = __expf(s[t][r] - mnew);
#pragma unroll
            for (int j = 0; j < 4; ++j) o[j][r] *= alpha;
            lacc[r] *= alpha;
        }

        // ---- P: C-layout -> LDS -> A-layout (per-wave tile, no barrier) ----
#pragma unroll
        for (int t = 0; t < 4; ++t)
#pragma unroll
            for (int r = 0; r < 8; ++r)
                Plds[w][r + 8 * half][t * 16 + ln] = (__bf16)s[t][r];

        Frag pf[2];
        const __bf16* prow = &Plds[w][ln][0];
        load_a_frag(pf[0], prow, 0, half);   // keys 0..31
        load_a_frag(pf[1], prow, 32, half);  // keys 32..63

        // ---- denominator via matrix core: l = l*alpha + P @ 1 ----
        lacc = wmma_bf16(pf[0].v, ones.v, lacc);
        lacc = wmma_bf16(pf[1].v, ones.v, lacc);

        // ---- O += P @ V  via V^T rows (contiguous along keys) ----
#pragma unroll
        for (int j = 0; j < 4; ++j) {
            const __bf16* vrow = &Vlds[buf][j * 16 + ln][0];
            Frag vb0, vb1;
            load_b_fragT(vb0, vrow, 0, half);
            load_b_fragT(vb1, vrow, 32, half);
            o[j] = wmma_bf16(pf[0].v, vb0.v, o[j]);
            o[j] = wmma_bf16(pf[1].v, vb1.v, o[j]);
        }
        __syncthreads();  // all reads of buf done before it is overwritten
    }

    // ---- epilogue: O / l  -> Ob in (B, N, D) bf16 row-major ----
    const int bb = bh >> 4, h = bh & 15;
#pragma unroll
    for (int r = 0; r < 8; ++r) {
        const float inv = 1.0f / lacc[r];
        const int token = qbase + r + 8 * half;
        __bf16* orow = Ob + ((size_t)(bb * 2048 + token)) * 1024 + h * 64;
#pragma unroll
        for (int j = 0; j < 4; ++j)
            orow[j * 16 + ln] = (__bf16)(o[j][r] * inv);
    }
}

// ---------------------------------------------------------------------------
// Host-side orchestration.
// ---------------------------------------------------------------------------
extern "C" void kernel_launch(void* const* d_in, const int* in_sizes, int n_in,
                              void* d_out, int out_size, void* d_ws,
                              size_t ws_size, hipStream_t stream) {
    (void)in_sizes; (void)n_in; (void)out_size; (void)ws_size;
    const float* x      = (const float*)d_in[0];  // (2, 2048, 1024)
    const float* W_qkv  = (const float*)d_in[1];  // (1024, 3072)
    const float* b_qkv  = (const float*)d_in[2];  // (3072,)
    const float* W_proj = (const float*)d_in[3];  // (1024, 1024)
    const float* b_proj = (const float*)d_in[4];  // (1024,)
    float* out = (float*)d_out;                   // (2, 2048, 1024)

    char* ws = (char*)d_ws;                       // 48 MiB total
    __bf16* xb     = (__bf16*)(ws);               // 4096x1024        (8 MiB)
    __bf16* WqkvT  = (__bf16*)(ws + 8388608);     // 3072x1024        (6 MiB)
    __bf16* WprojT = (__bf16*)(ws + 14680064);    // 1024x1024        (2 MiB)
    __bf16* Qb     = (__bf16*)(ws + 16777216);    // (32,2048,64)     (8 MiB)
    __bf16* Kb     = (__bf16*)(ws + 25165824);    // (32,2048,64)     (8 MiB)
    __bf16* Vtb    = (__bf16*)(ws + 33554432);    // (32,64,2048)     (8 MiB)
    __bf16* Ob     = (__bf16*)(ws + 41943040);    // 4096x1024        (8 MiB)

    cvt_bf16_kernel<<<(4194304 + 255) / 256, 256, 0, stream>>>(x, xb, 4194304);
    transpose_cvt_kernel<<<(3145728 + 255) / 256, 256, 0, stream>>>(
        W_qkv, WqkvT, 1024, 3072);
    transpose_cvt_kernel<<<(1048576 + 255) / 256, 256, 0, stream>>>(
        W_proj, WprojT, 1024, 1024);

    // QKV projection + head split (Q pre-scaled, V transposed)
    gemm_bf16_wmma<1><<<dim3(3072 / 128, 4096 / 128), 256, 0, stream>>>(
        xb, WqkvT, b_qkv, nullptr, Qb, Kb, Vtb, 4096, 3072, 1024);

    // Flash attention: 32 (b,h) pairs x 16 query blocks
    flash_attn_wmma<<<512, 256, 0, stream>>>(Qb, Kb, Vtb, Ob);

    // Output projection (fp32 out + bias)
    gemm_bf16_wmma<0><<<dim3(1024 / 128, 4096 / 128), 256, 0, stream>>>(
        Ob, WprojT, b_proj, out, nullptr, nullptr, nullptr, 4096, 1024, 1024);
}
